// SparseConv1x1_1915555414387
// MI455X (gfx1250) — matrix operand order repaired and checked by reading, not compiled, not run on hardware
//
#include <hip/hip_runtime.h>

typedef __attribute__((ext_vector_type(8)))  __bf16 v8bf;
typedef __attribute__((ext_vector_type(16))) __bf16 v16bf;
typedef __attribute__((ext_vector_type(8)))  float  v8f;

#define IN_C    1024
#define OUT_C   1024
#define HW      784           // 28*28
#define PTILES  49            // 784/16 column tiles per sample
#define P_TILE  16
#define T_BLOCK 4             // p-tiles per block -> 64 output columns
#define M_BLOCK 128           // 8 waves x 16 rows
#define K_TILE  64            // staged K per barrier = 2 WMMA substeps of 32
#define KP      (K_TILE + 8)  // padded LDS row stride (bf16): 144B = 36 dwords,
                              // 36*r mod 64 distinct for r=0..15 -> conflict-free
#define NK      (IN_C / K_TILE)

// Stage one 64-k slab (double-buffered): W tile 128x64 and 4 x-tiles 16x64,
// f32 -> bf16 in registers; every LDS store is a b128.
__device__ __forceinline__ void stage_tiles(
    __bf16 (* __restrict__ A)[KP],              // [128][KP]
    __bf16 (* __restrict__ B)[P_TILE][KP],      // [4][16][KP]
    const float* __restrict__ w,
    const float* __restrict__ x,
    const size_t* xcol,                          // per-tile base: n*IN_C*HW + p0
    int m0, int k0, int tid)
{
    // ---- weight tile: thread = (row, 32-k half); 128B contiguous global read ----
    const int r  = tid >> 1;                     // 0..127
    const int kh = (tid & 1) * 32;               // 0 or 32
    const float* wrow = w + (size_t)(m0 + r) * IN_C + k0 + kh;
    __builtin_prefetch(wrow + K_TILE, 0, 1);     // global_prefetch_b8, next slab
    #pragma unroll
    for (int i = 0; i < 32; i += 8) {
        __bf16 t8[8];
        #pragma unroll
        for (int j = 0; j < 8; ++j) t8[j] = (__bf16)wrow[i + j];
        *(v8bf*)&A[r][kh + i] = *(const v8bf*)t8;
    }

    // ---- x tiles, transposed in registers: thread gathers 16 k for fixed (tile,p).
    // Lanes vary p fastest -> each strided load coalesces across the wave.
    const int kg = tid >> 6;                     // 0..3 (16-k group)
    const int tt = (tid >> 4) & 3;               // tile 0..3
    const int p  = tid & 15;
    const float* xb = x + xcol[tt] + p;
    #pragma unroll
    for (int h = 0; h < 2; ++h) {
        __bf16 t8[8];
        #pragma unroll
        for (int j = 0; j < 8; ++j)
            t8[j] = (__bf16)xb[(size_t)(k0 + kg * 16 + h * 8 + j) * HW];
        *(v8bf*)&B[tt][p][kg * 16 + h * 8] = *(const v8bf*)t8;
    }
}

__global__ __launch_bounds__(256)
void conv1x1_wmma_bf16(const float* __restrict__ x,
                       const float* __restrict__ w,
                       const float* __restrict__ bias,
                       float* __restrict__ out,
                       int ntiles)
{
    __shared__ __bf16 As[2][M_BLOCK][KP];            // 36 KB
    __shared__ __bf16 Bs[2][T_BLOCK][P_TILE][KP];    // 18 KB

    const int tid  = threadIdx.x;
    const int wv   = tid >> 5;       // wave 0..7 (wave32)
    const int lane = tid & 31;
    const int half = lane >> 4;
    const int l16  = lane & 15;
    const int m0   = blockIdx.y * M_BLOCK;

    int    tg[T_BLOCK];
    size_t xcol[T_BLOCK];
    #pragma unroll
    for (int t = 0; t < T_BLOCK; ++t) {
        tg[t] = blockIdx.x * T_BLOCK + t;
        const int tgc = tg[t] < ntiles ? tg[t] : ntiles - 1;   // clamp (safety)
        const int n   = tgc / PTILES;
        const int p0  = (tgc % PTILES) * P_TILE;
        xcol[t] = (size_t)n * IN_C * HW + p0;
    }

    v8f acc[T_BLOCK] = {};           // 4 x f32 16x16 accumulators

    stage_tiles(As[0], Bs[0], w, x, xcol, m0, 0, tid);
    __syncthreads();

    for (int kt = 0; kt < NK; ++kt) {
        const int cur = kt & 1;
        // overlap: stage slab kt+1 while computing slab kt
        if (kt + 1 < NK)
            stage_tiles(As[cur ^ 1], Bs[cur ^ 1], w, x, xcol,
                        m0, (kt + 1) * K_TILE, tid);

        #pragma unroll
        for (int ks = 0; ks < 2; ++ks) {         // two 32-k WMMA substeps
            // A fragment (16-bit 16x32 layout): lane m = l16,
            // VGPR0-7: k = half*16..+15 (contiguous, matches B)
            const __bf16* ap = &As[cur][wv * 16 + l16][ks * 32 + half * 16];
            union { v16bf v; v8bf h[2]; } au;
            au.h[0] = *(const v8bf*)ap;          // 16B-aligned, conflict-free
            au.h[1] = *(const v8bf*)(ap + 8);

            // batch all 4 B fragments first -> single dscnt wait, then
            // 4 back-to-back WMMAs reusing one A fragment
            v16bf bfrag[T_BLOCK];
            #pragma unroll
            for (int t = 0; t < T_BLOCK; ++t)
                bfrag[t] = *(const v16bf*)&Bs[cur][t][l16][ks * 32 + half * 16];

            #pragma unroll
            for (int t = 0; t < T_BLOCK; ++t)
                acc[t] = __builtin_amdgcn_wmma_f32_16x16x32_bf16(
                    false, au.v, false, bfrag[t], (short)0, acc[t], false, false);
        }
        __syncthreads();   // one barrier per 8 WMMAs; protects both buffers
    }

    // epilogue: C/D layout -> VGPR v: M = half*8 + v, N = l16
    const int row_base = m0 + wv * 16 + half * 8;
    float bv[8];
    #pragma unroll
    for (int v = 0; v < 8; ++v) bv[v] = bias[row_base + v];

    #pragma unroll
    for (int t = 0; t < T_BLOCK; ++t) {
        if (tg[t] >= ntiles) continue;
        const int n  = tg[t] / PTILES;
        const int p0 = (tg[t] % PTILES) * P_TILE;
        float* op = out + ((size_t)n * OUT_C + row_base) * HW + p0 + l16;
        #pragma unroll
        for (int v = 0; v < 8; ++v)
            op[(size_t)v * HW] = acc[t][v] + bv[v];
    }
}

extern "C" void kernel_launch(void* const* d_in, const int* in_sizes, int n_in,
                              void* d_out, int out_size, void* d_ws, size_t ws_size,
                              hipStream_t stream) {
    const float* x    = (const float*)d_in[0];   // [N, 1024, 28, 28] f32
    const float* w    = (const float*)d_in[1];   // [1024, 1024] f32
    const float* bias = (const float*)d_in[2];   // [1024, 1] f32
    float* out        = (float*)d_out;           // [N, 1024, 28, 28] f32

    const int nsamples = in_sizes[0] / (IN_C * HW);   // 32
    const int ntiles   = nsamples * PTILES;           // 1568
    dim3 grid((ntiles + T_BLOCK - 1) / T_BLOCK,       // 392
              OUT_C / M_BLOCK);                       // 8
    conv1x1_wmma_bf16<<<grid, dim3(256, 1, 1), 0, stream>>>(x, w, bias, out, ntiles);
}
